// Block_70093866270826
// MI455X (gfx1250) — matrix operand order repaired and checked by reading, not compiled, not run on hardware
//
#include <hip/hip_runtime.h>
#include <cstdint>
#include <cstddef>

// ---- types for CDNA5 WMMA (wave32) ----
typedef __bf16 bf16;
typedef bf16  v16bf __attribute__((ext_vector_type(16)));
typedef bf16  v8bf  __attribute__((ext_vector_type(8)));
typedef float v8f   __attribute__((ext_vector_type(8)));

union Frag { v16bf v; v8bf h[2]; };

#define NB   2048      // B*T tokens
#define E_   1024
#define H_   8
#define D_   63
#define HD   504       // H*D
#define QKV_PAD 1536   // 3*H*D=1512 padded to multiple of 128
#define PROJ_K  512    // H*D=504 padded to multiple of 32
#define FF   4096

// CDNA5 async global->LDS copy: 16 bytes per lane, tracked by ASYNCcnt.
__device__ __forceinline__ void async_cp16(uint32_t lds_addr, const bf16* gptr) {
  asm volatile("global_load_async_to_lds_b128 %0, %1, off"
               :: "v"(lds_addr), "v"(gptr) : "memory");
}

// ---------------- weight fp32 -> bf16 with zero padding ----------------
__global__ void convert_w_kernel(const float* __restrict__ in, bf16* __restrict__ out,
                                 int rin, int cin, int rout, int cout) {
  int idx = blockIdx.x * blockDim.x + threadIdx.x;
  int total = rout * cout;
  if (idx >= total) return;
  int r = idx / cout, c = idx - r * cout;
  float v = (r < rin && c < cin) ? in[(size_t)r * cin + c] : 0.0f;
  out[idx] = (bf16)v;
}

// ---------------- rotary embedding + LayerNorm1 ----------------
__global__ __launch_bounds__(256) void rotary_ln1_kernel(
    const float* __restrict__ x, const float* __restrict__ w, const float* __restrict__ b,
    float* __restrict__ xr, bf16* __restrict__ h1) {
  const int n = blockIdx.x;
  const int tid = threadIdx.x;
  const float* xrow = x + (size_t)n * 512;
  float vals[4];
  int cols[4];
  float sum = 0.f, sq = 0.f;
  const float lf = 9.210340371976184f / 1024.0f; // log(10000)/dim
  #pragma unroll
  for (int i = 0; i < 2; ++i) {
    int j = tid + 256 * i;
    float invf = __expf(-(float)(2 * j) * lf);
    float a = xrow[j] * invf;
    float s, c;
    __sincosf(a, &s, &c);
    vals[2 * i] = s;     cols[2 * i] = j;
    vals[2 * i + 1] = c; cols[2 * i + 1] = 512 + j;
    sum += s + c; sq += s * s + c * c;
  }
  __shared__ float rs[256], rq[256];
  rs[tid] = sum; rq[tid] = sq; __syncthreads();
  for (int s = 128; s > 0; s >>= 1) {
    if (tid < s) { rs[tid] += rs[tid + s]; rq[tid] += rq[tid + s]; }
    __syncthreads();
  }
  float mu = rs[0] * (1.0f / 1024.0f);
  float var = rq[0] * (1.0f / 1024.0f) - mu * mu;
  float rstd = rsqrtf(var + 1e-5f);
  #pragma unroll
  for (int i = 0; i < 4; ++i) {
    int c = cols[i];
    float v = vals[i];
    xr[(size_t)n * E_ + c] = v;
    h1[(size_t)n * E_ + c] = (bf16)((v - mu) * rstd * w[c] + b[c]);
  }
}

// ---------------- LayerNorm2 ----------------
__global__ __launch_bounds__(256) void ln2_kernel(
    const float* __restrict__ xa, const float* __restrict__ w, const float* __restrict__ b,
    bf16* __restrict__ h2) {
  const int n = blockIdx.x;
  const int tid = threadIdx.x;
  const float* row = xa + (size_t)n * E_;
  float vals[4];
  float sum = 0.f, sq = 0.f;
  #pragma unroll
  for (int i = 0; i < 4; ++i) {
    float v = row[tid + 256 * i];
    vals[i] = v; sum += v; sq += v * v;
  }
  __shared__ float rs[256], rq[256];
  rs[tid] = sum; rq[tid] = sq; __syncthreads();
  for (int s = 128; s > 0; s >>= 1) {
    if (tid < s) { rs[tid] += rs[tid + s]; rq[tid] += rq[tid + s]; }
    __syncthreads();
  }
  float mu = rs[0] * (1.0f / 1024.0f);
  float var = rq[0] * (1.0f / 1024.0f) - mu * mu;
  float rstd = rsqrtf(var + 1e-5f);
  #pragma unroll
  for (int i = 0; i < 4; ++i) {
    int c = tid + 256 * i;
    h2[(size_t)n * E_ + c] = (bf16)((vals[i] - mu) * rstd * w[c] + b[c]);
  }
}

// ---------------- WMMA GEMM with async-LDS double buffering ----------------
// C[M,Nout] = A[M,K](bf16) x W[Nout,K]^T(bf16), K % 32 == 0, M % 128 == 0, Nout % 128 == 0.
// Block = 256 threads (8 waves), block tile 128x128; wave tile 32x64 (8 WMMAs/K-step).
// Per K-step: A(128x32) and B(128x32) bf16 tiles staged to LDS by
// global_load_async_to_lds_b128 (4 per thread), double buffered; fragments read
// back with ds_load_b128 in the CDNA5 16-bit A/B register layouts.
// mode 0: C = acc + bias (fp32) | mode 1: + res (fp32) | mode 2: gelu -> bf16
__global__ __launch_bounds__(256) void gemm_wmma_kernel(
    const bf16* __restrict__ A, const bf16* __restrict__ W,
    const float* __restrict__ bias, const float* __restrict__ res,
    float* __restrict__ Cf, bf16* __restrict__ Cb,
    int K, int ldc, int nreal, int mode) {
  __shared__ bf16 sA[2][128 * 32];
  __shared__ bf16 sB[2][128 * 32];
  const int t = threadIdx.x;
  const int lane = t & 31;
  const int wave = t >> 5;
  const int mw = wave & 3;          // M sub-block (32 rows)
  const int nw = wave >> 2;         // N sub-block (64 cols)
  const int mblk = blockIdx.x * 128;
  const int nblk = blockIdx.y * 128;
  const int half = lane >> 4;       // 0: lanes 0-15, 1: lanes 16-31
  const int lm = lane & 15;

  // cooperative staging: two 16B chunks of A and two of B per thread
  const int c0 = t, c1 = t + 256;
  const int ar0 = c0 >> 2, ac0 = (c0 & 3) * 8;   // row 0..127, col offset in elems
  const int ar1 = c1 >> 2, ac1 = (c1 & 3) * 8;
  const bf16* gA0 = A + (size_t)(mblk + ar0) * K + ac0;
  const bf16* gA1 = A + (size_t)(mblk + ar1) * K + ac1;
  const bf16* gB0 = W + (size_t)(nblk + ar0) * K + ac0;
  const bf16* gB1 = W + (size_t)(nblk + ar1) * K + ac1;
  const uint32_t lA0 = (uint32_t)(uintptr_t)(&sA[0][ar0 * 32 + ac0]);
  const uint32_t lA1 = (uint32_t)(uintptr_t)(&sA[0][ar1 * 32 + ac1]);
  const uint32_t lB0 = (uint32_t)(uintptr_t)(&sB[0][ar0 * 32 + ac0]);
  const uint32_t lB1 = (uint32_t)(uintptr_t)(&sB[0][ar1 * 32 + ac1]);
  const uint32_t bufoff = 128 * 32 * 2;          // bytes between double buffers

  v8f acc[2][4] = {};
  const int nk = K >> 5;

  // prologue: stage K-step 0 into buffer 0
  async_cp16(lA0, gA0); async_cp16(lA1, gA1);
  async_cp16(lB0, gB0); async_cp16(lB1, gB1);

  for (int kb = 0; kb < nk; ++kb) {
    const int cur = kb & 1;
    if (kb + 1 < nk) {
      const uint32_t o = ((kb + 1) & 1) ? bufoff : 0u;
      const size_t gofs = (size_t)(kb + 1) * 32;
      async_cp16(lA0 + o, gA0 + gofs); async_cp16(lA1 + o, gA1 + gofs);
      async_cp16(lB0 + o, gB0 + gofs); async_cp16(lB1 + o, gB1 + gofs);
      // current buffer's 4 async ops done; next buffer's 4 may remain in flight
      asm volatile("s_wait_asynccnt 4" ::: "memory");
    } else {
      asm volatile("s_wait_asynccnt 0" ::: "memory");
    }
    __syncthreads();

    const bf16* baseA = &sA[cur][0];
    const bf16* baseB = &sB[cur][0];
    Frag a[2], b[4];
    #pragma unroll
    for (int ms = 0; ms < 2; ++ms) {
      const bf16* pa = baseA + (mw * 32 + ms * 16 + lm) * 32 + half * 8;
      a[ms].h[0] = *(const v8bf*)(pa);
      a[ms].h[1] = *(const v8bf*)(pa + 16);
    }
    #pragma unroll
    for (int s = 0; s < 4; ++s) {
      const bf16* pb = baseB + (nw * 64 + 16 * s + lm) * 32 + half * 8;
      b[s].h[0] = *(const v8bf*)(pb);
      b[s].h[1] = *(const v8bf*)(pb + 16);
    }
    #pragma unroll
    for (int ms = 0; ms < 2; ++ms)
      #pragma unroll
      for (int s = 0; s < 4; ++s)
        acc[ms][s] = __builtin_amdgcn_wmma_f32_16x16x32_bf16(
            false, a[ms].v, false, b[s].v, (short)0, acc[ms][s], false, false);
    __syncthreads();   // all reads of buf[cur] done before it is refilled
  }

  // epilogue: lane holds col = nblk + nw*64 + 16*s + lm, rows + half*8 + r
  #pragma unroll
  for (int ms = 0; ms < 2; ++ms) {
    #pragma unroll
    for (int s = 0; s < 4; ++s) {
      int col = nblk + nw * 64 + 16 * s + lm;
      float bv = (col < nreal) ? bias[col] : 0.0f;
      #pragma unroll
      for (int r = 0; r < 8; ++r) {
        int row = mblk + mw * 32 + ms * 16 + half * 8 + r;
        float v = acc[ms][s][r] + bv;
        if (mode == 1) v += res[(size_t)row * ldc + col];
        if (mode == 2) {
          float g = 0.5f * v * (1.0f + erff(v * 0.7071067811865476f));
          Cb[(size_t)row * ldc + col] = (bf16)g;
        } else {
          Cf[(size_t)row * ldc + col] = v;
        }
      }
    }
  }
}

// ---------------- windowed attention: per (token, head) 63x63 softmax ----------------
__global__ __launch_bounds__(64) void attn_kernel(
    const float* __restrict__ qkv, const float* __restrict__ rel_pos,
    bf16* __restrict__ outp) {
  const int unit = blockIdx.x;
  const int n = unit >> 3, h = unit & 7;
  const float* qp = qkv + (size_t)n * QKV_PAD + h * D_;
  const float* kp = qp + HD;
  const float* vp = qp + 2 * HD;
  __shared__ float S[D_ * D_];
  __shared__ float sc[64];
  __shared__ float qs[D_];
  const int v = threadIdx.x;
  if (v < D_) qs[v] = qp[v];
  __syncthreads();
  if (v < D_) {
    float kv = kp[v], vv = vp[v];
    float denom = 0.f;
    for (int d = 0; d < D_; ++d) {
      int br = d + 32; if (br >= D_) br -= D_;  // (d-31) mod 63
      float e = __expf(qs[d] * kv + rel_pos[br * D_ + v]);
      S[d * D_ + v] = e;
      denom += e;
    }
    sc[v] = vv / denom;
  }
  __syncthreads();
  if (v < D_) {
    float acc = 0.f;
    for (int j = 0; j < D_; ++j) acc += S[v * D_ + j] * sc[j];
    outp[(size_t)n * PROJ_K + h * D_ + v] = (bf16)acc;
  }
  if (h == 0 && v >= 56) {  // zero the K-padding cols 504..511
    outp[(size_t)n * PROJ_K + HD + (v - 56)] = (bf16)0.0f;
  }
}

extern "C" void kernel_launch(void* const* d_in, const int* in_sizes, int n_in,
                              void* d_out, int out_size, void* d_ws, size_t ws_size,
                              hipStream_t stream) {
  const float* x       = (const float*)d_in[0];
  const float* ln1_w   = (const float*)d_in[1];
  const float* ln1_b   = (const float*)d_in[2];
  const float* qkv_w   = (const float*)d_in[3];
  const float* qkv_b   = (const float*)d_in[4];
  const float* rel_pos = (const float*)d_in[5];
  const float* proj_w  = (const float*)d_in[6];
  const float* proj_b  = (const float*)d_in[7];
  const float* ln2_w   = (const float*)d_in[8];
  const float* ln2_b   = (const float*)d_in[9];
  const float* fc_w    = (const float*)d_in[10];
  const float* fc_b    = (const float*)d_in[11];
  const float* cproj_w = (const float*)d_in[12];
  const float* cproj_b = (const float*)d_in[13];

  char* ws = (char*)d_ws;
  size_t off = 0;
  auto take = [&](size_t bytes) -> char* {
    char* p = ws + off;
    off = (off + bytes + 255) & ~(size_t)255;
    return p;
  };
  float* xr    = (float*)take((size_t)NB * E_ * 4);        // rotary output (residual)
  bf16*  h1    = (bf16*) take((size_t)NB * E_ * 2);        // LN1 out
  bf16*  Wqkv  = (bf16*) take((size_t)QKV_PAD * E_ * 2);   // padded bf16 weights
  bf16*  Wproj = (bf16*) take((size_t)E_ * PROJ_K * 2);
  bf16*  Wfc   = (bf16*) take((size_t)FF * E_ * 2);
  bf16*  Wcp   = (bf16*) take((size_t)E_ * FF * 2);
  float* qkv   = (float*)take((size_t)NB * QKV_PAD * 4);   // qkv activations
  bf16*  ao    = (bf16*) take((size_t)NB * PROJ_K * 2);    // attention out (padded)
  float* xa    = (float*)take((size_t)NB * E_ * 4);        // residual after attention
  bf16*  h2    = (bf16*) take((size_t)NB * E_ * 2);        // LN2 out
  bf16*  g     = (bf16*) take((size_t)NB * FF * 2);        // gelu(fc) out

  auto cvt = [&](const float* in, bf16* out, int rin, int cin, int rout, int cout) {
    int total = rout * cout;
    convert_w_kernel<<<(total + 255) / 256, 256, 0, stream>>>(in, out, rin, cin, rout, cout);
  };
  cvt(qkv_w,   Wqkv,  1512, E_, QKV_PAD, E_);
  cvt(proj_w,  Wproj, E_,   HD, E_,      PROJ_K);
  cvt(fc_w,    Wfc,   FF,   E_, FF,      E_);
  cvt(cproj_w, Wcp,   E_,   FF, E_,      FF);

  rotary_ln1_kernel<<<NB, 256, 0, stream>>>(x, ln1_w, ln1_b, xr, h1);

  // qkv = h1 @ qkv_w^T + qkv_b   -> fp32 [NB, 1536]
  gemm_wmma_kernel<<<dim3(NB / 128, QKV_PAD / 128), 256, 0, stream>>>(
      h1, Wqkv, qkv_b, nullptr, qkv, nullptr, E_, QKV_PAD, 1512, 0);

  attn_kernel<<<NB * H_, 64, 0, stream>>>(qkv, rel_pos, ao);

  // xa = xr + ao @ proj_w^T + proj_b
  gemm_wmma_kernel<<<dim3(NB / 128, E_ / 128), 256, 0, stream>>>(
      ao, Wproj, proj_b, xr, xa, nullptr, PROJ_K, E_, E_, 1);

  ln2_kernel<<<NB, 256, 0, stream>>>(xa, ln2_w, ln2_b, h2);

  // g = gelu(h2 @ fc_w^T + fc_b)  -> bf16 [NB, 4096]
  gemm_wmma_kernel<<<dim3(NB / 128, FF / 128), 256, 0, stream>>>(
      h2, Wfc, fc_b, nullptr, nullptr, g, E_, FF, FF, 2);

  // out = xa + g @ cproj_w^T + cproj_b
  gemm_wmma_kernel<<<dim3(NB / 128, E_ / 128), 256, 0, stream>>>(
      g, Wcp, cproj_b, xa, (float*)d_out, nullptr, FF, E_, E_, 1);
}
